// AML_GIN_78769700208815
// MI455X (gfx1250) — matrix operand order repaired and checked by reading, not compiled, not run on hardware
//
#include <hip/hip_runtime.h>
#include <hip/hip_bf16.h>

// ---------------------------------------------------------------------------
// GIN forward for MI455X (gfx1250): wave32, WMMA f32_16x16x32_f16.
// Fully fused GIN-MLP kernel: (h+agg)->f16 -> GEMM1(relu) -> GEMM2 -> BN stats,
// all staged through 8KB of LDS per block; weights pre-shuffled into WMMA
// B-fragment order so every operand load is a b128.
// ---------------------------------------------------------------------------

#define NNODES 100000
#define NEDGES 1600000
#define HDIM   64
#define BN_EPS 1e-5f

typedef _Float16 f16;
typedef __attribute__((ext_vector_type(16))) _Float16 v16h;
typedef __attribute__((ext_vector_type(8)))  _Float16 v8h;
typedef __attribute__((ext_vector_type(4)))  _Float16 v4h;
typedef __attribute__((ext_vector_type(8)))  float    v8f;
typedef __attribute__((ext_vector_type(4)))  float    v4f;

// ---------------- generic zero ----------------
__global__ void k_zero_f32(float* __restrict__ p, long n) {
    long i = (long)blockIdx.x * blockDim.x + threadIdx.x;
    long stride = (long)gridDim.x * blockDim.x;
    for (; i < n; i += stride) p[i] = 0.0f;
}

// ---------------- edge scatter-mean (edge_attr -> esum, counts) ------------
__global__ void k_edge_scatter(const float* __restrict__ ea,
                               const int* __restrict__ ei,
                               float* __restrict__ esum,
                               float* __restrict__ cnt) {
    long tid = (long)blockIdx.x * blockDim.x + threadIdx.x;
    if (tid >= (long)NEDGES * 4) return;
    int e = (int)(tid >> 2);
    int q = (int)(tid & 3);
    int r = ei[e];                       // row = edge_index[0][e]
    float4 v = *(const float4*)(ea + (long)e * 16 + q * 4);
    float* dst = esum + (long)r * 16 + q * 4;
    atomicAdd(dst + 0, v.x); atomicAdd(dst + 1, v.y);
    atomicAdd(dst + 2, v.z); atomicAdd(dst + 3, v.w);
    if (q == 0) atomicAdd(&cnt[r], 1.0f);
}

// ---------------- h = concat(x, esum/max(cnt,1), zeros) --------------------
__global__ void k_build_h(const float* __restrict__ x,
                          const float* __restrict__ esum,
                          const float* __restrict__ cnt,
                          float* __restrict__ h) {
    long tid = (long)blockIdx.x * blockDim.x + threadIdx.x;
    if (tid >= (long)NNODES * HDIM) return;
    int i = (int)(tid >> 6);
    int c = (int)(tid & 63);
    float v;
    if (c < 32)       v = x[(long)i * 32 + c];
    else if (c < 48)  v = esum[(long)i * 16 + (c - 32)] / fmaxf(cnt[i], 1.0f);
    else              v = 0.0f;
    h[tid] = v;
}

// ---------------- GIN message scatter: agg[col] += h[row] ------------------
__global__ void k_gin_scatter(const int* __restrict__ ei,
                              const float* __restrict__ h,
                              float* __restrict__ agg) {
    long tid = (long)blockIdx.x * blockDim.x + threadIdx.x;
    if (tid >= (long)NEDGES * 16) return;
    int e = (int)(tid >> 4);
    int q = (int)(tid & 15);
    int r  = ei[e];              // source
    int co = ei[NEDGES + e];     // target
    float4 v = *(const float4*)(h + (long)r * HDIM + q * 4);
    float* dst = agg + (long)co * HDIM + q * 4;
    atomicAdd(dst + 0, v.x); atomicAdd(dst + 1, v.y);
    atomicAdd(dst + 2, v.z); atomicAdd(dst + 3, v.w);
}

// ---------------- fp32 weight -> fp16 in WMMA B-fragment order -------------
// (ISA 7.12.2, 32x16 f16 B): lane col = lane&15, K = (lane>>4)*8 + (j&7) + (j>>3)*16.
// wf[((n0t*2 + kt)*32 + lane)*16 + j]
__global__ void k_conv_wfrag(const float* __restrict__ w, f16* __restrict__ wf,
                             int rows) {
    int tid = blockIdx.x * blockDim.x + threadIdx.x;
    if (tid >= 4096) return;
    int n0t  = tid >> 10;          // output col tile 0..3
    int kt   = (tid >> 9) & 1;     // k tile 0..1
    int lane = (tid >> 4) & 31;    // lane 0..31
    int j    = tid & 15;
    int row = kt * 32 + (lane >> 4) * 8 + (j & 7) + (j >> 3) * 16;
    int col = n0t * 16 + (lane & 15);
    float v = (row < rows) ? w[row * HDIM + col] : 0.0f;
    wf[(((n0t * 2 + kt) << 5) + lane) * 16 + j] = (f16)v;
}

// ---------------- fused GIN MLP ---------------------------------------------
// One block = one 16-row tile; 4 waves cover the 64 output columns.
//   stage 1: z   = f16(h + agg)            (global float4 -> LDS v4h)
//   stage 2: t   = relu(z @ W1 + b1)       (2 WMMA per wave, A from LDS)
//   stage 3: z2  = t @ W2 + b2             (2 WMMA per wave, A from LDS)
//   stage 4: store z2 (coalesced b128) + per-block BN sum/sumsq -> atomics
__global__ void k_gin_mlp(const float* __restrict__ h,
                          const float* __restrict__ agg,
                          const f16* __restrict__ Wf1, const float* __restrict__ b1,
                          const f16* __restrict__ Wf2, const float* __restrict__ b2,
                          float* __restrict__ z2out, float* __restrict__ stats) {
    __shared__ __align__(32) unsigned char smem[8192];
    f16*   zs  = (f16*)smem;             // [16][64] f16 : z
    f16*   ts  = (f16*)(smem + 2048);    // [16][64] f16 : relu hidden
    float* t32 = (float*)(smem + 4096);  // [16][64] f32 : z2

    const int tid = threadIdx.x;         // 128 threads
    const int m0  = blockIdx.x * 16;

    // ---- stage 1: z = f16(h + agg) into LDS ----
    {
        const float4* hp = (const float4*)(h   + (size_t)m0 * HDIM);
        const float4* gp = (const float4*)(agg + (size_t)m0 * HDIM);
#pragma unroll
        for (int k = 0; k < 2; ++k) {
            int idx = tid * 2 + k;                 // 256 float4 chunks
            float4 a = hp[idx];
            float4 g = gp[idx];
            v4h o;
            o[0] = (f16)(a.x + g.x); o[1] = (f16)(a.y + g.y);
            o[2] = (f16)(a.z + g.z); o[3] = (f16)(a.w + g.w);
            ((v4h*)zs)[idx] = o;
        }
    }
    __syncthreads();

    const int lane = tid & 31;
    const int wv   = tid >> 5;
    const int n0   = wv * 16;
    const int r    = lane & 15;          // A row within tile
    const int hi   = lane >> 4;
    const int khA  = hi * 8;             // A K-half offset
    const int cN   = lane & 15;          // D column within tile

    // ---- stage 2: t = relu(z @ W1 + b1) ----
    {
        const float bv = b1[n0 + cN];
        v8f acc;
#pragma unroll
        for (int g = 0; g < 8; ++g) acc[g] = bv;
        const f16* wbase = Wf1 + ((((size_t)wv * 2) << 5) | lane) * 16;
#pragma unroll
        for (int kt = 0; kt < 2; ++kt) {
            union { v16h v; v8h hh[2]; } a;
            const f16* arow = zs + r * HDIM + kt * 32 + khA;
            a.hh[0] = *(const v8h*)(arow);         // K = kt*32 + khA + 0..7
            a.hh[1] = *(const v8h*)(arow + 16);    // K = kt*32 + 16 + khA + 0..7
            v16h b = *(const v16h*)(wbase + (size_t)kt * 512);
            acc = __builtin_amdgcn_wmma_f32_16x16x32_f16(
                false, a.v, false, b, (short)0, acc, false, false);
        }
#pragma unroll
        for (int g = 0; g < 8; ++g)
            ts[(hi * 8 + g) * HDIM + n0 + cN] = (f16)fmaxf(acc[g], 0.0f);
    }
    __syncthreads();

    // ---- stage 3: z2 = t @ W2 + b2 ----
    {
        const float bv = b2[n0 + cN];
        v8f acc;
#pragma unroll
        for (int g = 0; g < 8; ++g) acc[g] = bv;
        const f16* wbase = Wf2 + ((((size_t)wv * 2) << 5) | lane) * 16;
#pragma unroll
        for (int kt = 0; kt < 2; ++kt) {
            union { v16h v; v8h hh[2]; } a;
            const f16* arow = ts + r * HDIM + kt * 32 + khA;
            a.hh[0] = *(const v8h*)(arow);
            a.hh[1] = *(const v8h*)(arow + 16);
            v16h b = *(const v16h*)(wbase + (size_t)kt * 512);
            acc = __builtin_amdgcn_wmma_f32_16x16x32_f16(
                false, a.v, false, b, (short)0, acc, false, false);
        }
#pragma unroll
        for (int g = 0; g < 8; ++g)
            t32[(hi * 8 + g) * HDIM + n0 + cN] = acc[g];
    }
    __syncthreads();

    // ---- stage 4: coalesced store of z2 + block-level BN stats ----
    {
        v4f* dst = (v4f*)(z2out + (size_t)m0 * HDIM);
        const v4f* src = (const v4f*)t32;
        dst[tid]       = src[tid];
        dst[tid + 128] = src[tid + 128];
        if (tid < HDIM) {
            float s = 0.0f, ss = 0.0f;
#pragma unroll
            for (int rr = 0; rr < 16; ++rr) {
                float v = t32[rr * HDIM + tid];
                s += v; ss += v * v;
            }
            atomicAdd(&stats[tid], s);
            atomicAdd(&stats[HDIM + tid], ss);
        }
    }
}

// ---------------- BN apply + ReLU -> h, vectorized x4 ----------------------
__global__ void k_bn_apply(const float* __restrict__ z,
                           const float* __restrict__ st,
                           const float* __restrict__ gam,
                           const float* __restrict__ bet,
                           float* __restrict__ h) {
    long tid = (long)blockIdx.x * blockDim.x + threadIdx.x;
    if (tid >= (long)NNODES * HDIM / 4) return;
    const int c0 = (int)((tid * 4) & 63);
    const float invN = 1.0f / (float)NNODES;
    float4 zv = ((const float4*)z)[tid];
    float in[4] = {zv.x, zv.y, zv.z, zv.w};
    float o[4];
#pragma unroll
    for (int j = 0; j < 4; ++j) {
        int c = c0 + j;
        float mean = st[c] * invN;
        float var  = st[HDIM + c] * invN - mean * mean;
        float rstd = rsqrtf(var + BN_EPS);
        o[j] = fmaxf((in[j] - mean) * rstd * gam[c] + bet[c], 0.0f);
    }
    float4 res = {o[0], o[1], o[2], o[3]};
    ((float4*)h)[tid] = res;
}

// ---------------- global mean pool: strided accumulation + few atomics -----
__global__ void k_pool(const float* __restrict__ h, float* __restrict__ pooled) {
    int c   = threadIdx.x & 63;
    int sub = threadIdx.x >> 6;
    long row = (long)blockIdx.x * 4 + sub;
    long stride = (long)gridDim.x * 4;
    float s = 0.0f;
    for (long i = row; i < NNODES; i += stride) s += h[i * HDIM + c];
    atomicAdd(&pooled[c], s);
}

// ---------------- classifier head: 64 -> 32 (relu) -> 2 --------------------
__global__ void k_classifier(const float* __restrict__ pooled,
                             const float* __restrict__ cw1,
                             const float* __restrict__ cb1,
                             const float* __restrict__ cw2,
                             const float* __restrict__ cb2,
                             float* __restrict__ out) {
    __shared__ float p[64];
    __shared__ float hid[32];
    int t = threadIdx.x;   // 64 threads
    p[t] = pooled[t] * (1.0f / (float)NNODES);
    __syncthreads();
    if (t < 32) {
        float s = cb1[t];
        for (int k = 0; k < 64; ++k) s += p[k] * cw1[k * 32 + t];
        hid[t] = fmaxf(s, 0.0f);
    }
    __syncthreads();
    if (t < 2) {
        float s = cb2[t];
        for (int j = 0; j < 32; ++j) s += hid[j] * cw2[j * 2 + t];
        out[t] = s;
    }
}

// ---------------------------------------------------------------------------
extern "C" void kernel_launch(void* const* d_in, const int* in_sizes, int n_in,
                              void* d_out, int out_size, void* d_ws, size_t ws_size,
                              hipStream_t stream) {
    (void)in_sizes; (void)n_in; (void)out_size; (void)ws_size;

    const float* x   = (const float*)d_in[0];
    const float* ea  = (const float*)d_in[1];
    const float* w1[3] = {(const float*)d_in[2],  (const float*)d_in[8],  (const float*)d_in[14]};
    const float* b1[3] = {(const float*)d_in[3],  (const float*)d_in[9],  (const float*)d_in[15]};
    const float* w2[3] = {(const float*)d_in[4],  (const float*)d_in[10], (const float*)d_in[16]};
    const float* b2[3] = {(const float*)d_in[5],  (const float*)d_in[11], (const float*)d_in[17]};
    const float* gm[3] = {(const float*)d_in[6],  (const float*)d_in[12], (const float*)d_in[18]};
    const float* bt[3] = {(const float*)d_in[7],  (const float*)d_in[13], (const float*)d_in[19]};
    const float* cw1 = (const float*)d_in[20];
    const float* cb1 = (const float*)d_in[21];
    const float* cw2 = (const float*)d_in[22];
    const float* cb2 = (const float*)d_in[23];
    const int*   ei  = (const int*)d_in[24];
    float* out = (float*)d_out;

    // ---- workspace carve-up (256B aligned) ----
    char* base = (char*)d_ws;
    size_t off = 0;
    auto carve = [&](size_t bytes) -> char* {
        char* p = base + off;
        off = (off + bytes + 255) & ~(size_t)255;
        return p;
    };
    float* h     = (float*)carve((size_t)NNODES * HDIM * 4);  // features
    float* agg   = (float*)carve((size_t)NNODES * HDIM * 4);  // esum / scatter / z2
    f16*   w16   = (f16*)  carve((size_t)6 * HDIM * HDIM * 2);// fp16 weights (frag order)
    float* cnt   = (float*)carve((size_t)NNODES * 4);
    float* stats = (float*)carve(2 * HDIM * 4);
    float* pooled= (float*)carve(HDIM * 4);

    const long NH  = (long)NNODES * HDIM;   // 6.4M
    const long E16 = (long)NEDGES * 16;     // 25.6M
    const long E4  = (long)NEDGES * 4;      // 6.4M
    const int T = 256;
    auto blks = [](long n, int t) { return (unsigned)((n + t - 1) / t); };

    // ---- fp16 weight prep in fragment order (w1_0 padded from 48 rows) ----
    k_conv_wfrag<<<16, 256, 0, stream>>>(w1[0], w16 + 0 * 4096, 48);
    k_conv_wfrag<<<16, 256, 0, stream>>>(w2[0], w16 + 1 * 4096, 64);
    k_conv_wfrag<<<16, 256, 0, stream>>>(w1[1], w16 + 2 * 4096, 64);
    k_conv_wfrag<<<16, 256, 0, stream>>>(w2[1], w16 + 3 * 4096, 64);
    k_conv_wfrag<<<16, 256, 0, stream>>>(w1[2], w16 + 4 * 4096, 64);
    k_conv_wfrag<<<16, 256, 0, stream>>>(w2[2], w16 + 5 * 4096, 64);

    // ---- edge scatter-mean + build h [N,64] ----
    k_zero_f32<<<2048, T, 0, stream>>>(agg, NH);          // esum lives in agg[0:N*16]
    k_zero_f32<<<512,  T, 0, stream>>>(cnt, NNODES);
    k_edge_scatter<<<blks(E4, T), T, 0, stream>>>(ea, ei, agg, cnt);
    k_build_h<<<blks(NH, T), T, 0, stream>>>(x, agg, cnt, h);

    // ---- 3 GIN layers ----
    for (int l = 0; l < 3; ++l) {
        k_zero_f32<<<2048, T, 0, stream>>>(agg, NH);
        k_zero_f32<<<1, 128, 0, stream>>>(stats, 2 * HDIM);
        k_gin_scatter<<<blks(E16, T), T, 0, stream>>>(ei, h, agg);
        // fused: z=f16(h+agg); t=relu(z@W1+b1); z2=t@W2+b2 (->agg); BN stats
        k_gin_mlp<<<NNODES / 16, 128, 0, stream>>>(
            h, agg, w16 + (2 * l + 0) * 4096, b1[l],
                    w16 + (2 * l + 1) * 4096, b2[l], agg, stats);
        // BatchNorm (batch stats) + ReLU -> h
        k_bn_apply<<<blks(NH / 4, T), T, 0, stream>>>(agg, stats, gm[l], bt[l], h);
    }

    // ---- global mean pool + classifier ----
    k_zero_f32<<<1, 64, 0, stream>>>(pooled, HDIM);
    k_pool<<<512, 256, 0, stream>>>(h, pooled);
    k_classifier<<<1, 64, 0, stream>>>(pooled, cw1, cb1, cw2, cb2, out);
}
